// MultiheadFeedForward_26800595927292
// MI455X (gfx1250) — compile-verified, matrix operand order checked by reading, not run on hardware
//
#include <hip/hip_runtime.h>

#define D_MODEL 2048
#define N_HEADS 16
#define HEAD_DIM 128
#define FF_DIM 512
#define M_BLK 128
#define F_CHUNK 128
#define TOKENS 16384

typedef __attribute__((ext_vector_type(16))) __bf16 v16bf;
typedef __attribute__((ext_vector_type(8)))  __bf16 v8bf;
typedef __attribute__((ext_vector_type(8)))  float  v8f;
typedef int vi4 __attribute__((vector_size(16)));  // matches builtin's V4i

union Frag { v16bf v; v8bf h[2]; };

#if __has_builtin(__builtin_amdgcn_global_load_async_to_lds_b128) && \
    __has_builtin(__builtin_amdgcn_s_wait_asynccnt)
#define USE_ASYNC 1
#else
#define USE_ASYNC 0
#endif

__device__ __forceinline__ v8f wmma_bf16(v16bf a, v16bf b, v8f c) {
  // D = A(16x32 bf16) * B(32x16 bf16) + C(16x16 f32)
  return __builtin_amdgcn_wmma_f32_16x16x32_bf16(false, a, false, b, (short)0, c,
                                                 false, false);
}

// 16-byte global -> LDS stage (async on CDNA5, tracked by ASYNCcnt)
__device__ __forceinline__ void stage16(const __bf16* g, __bf16* l) {
#if USE_ASYNC
  __builtin_amdgcn_global_load_async_to_lds_b128(
      (__attribute__((address_space(1))) vi4*)(void*)g,
      (__attribute__((address_space(3))) vi4*)(void*)l, /*offset=*/0,
      /*cpol=*/0);
#else
  *(v8bf*)l = *(const v8bf*)g;
#endif
}

__device__ __forceinline__ void wait_stage() {
#if USE_ASYNC
  __builtin_amdgcn_s_wait_asynccnt(0);
#endif
}

// ---------------------------------------------------------------------------
// Prep: fp32 -> bf16 weight conversion + transpose.
// W1t[h][f][d] = W1[h][d][f]   (B-fragments: lane%16 -> N column,
// W2t[h][d][f] = W2[h][f][d]    per-lane contiguous K runs of 8 halves)
// ---------------------------------------------------------------------------
__global__ __launch_bounds__(256) void cvt_w_kernel(
    const float* __restrict__ W1, const float* __restrict__ W2,
    __bf16* __restrict__ W1t, __bf16* __restrict__ W2t) {
  const int per_head = HEAD_DIM * FF_DIM;  // 65536
  const int total = N_HEADS * per_head;
  for (int i = blockIdx.x * blockDim.x + threadIdx.x; i < total;
       i += gridDim.x * blockDim.x) {
    int h = i / per_head, r = i % per_head;
    {  // W1t[h][f][d]
      int f = r / HEAD_DIM, d = r % HEAD_DIM;
      W1t[i] = (__bf16)W1[h * per_head + d * FF_DIM + f];
    }
    {  // W2t[h][d][f]
      int d = r / FF_DIM, f = r % FF_DIM;
      W2t[i] = (__bf16)W2[h * per_head + f * HEAD_DIM + d];
    }
  }
}

// ---------------------------------------------------------------------------
// Main: one block = (128 tokens) x (1 head), 512 threads = 16 waves.
// LDS (128KB): hA[128][128] bf16 | W2 chunk [128][128] | W1 chunk x2 (dbl buf)
// FF dim processed in 4 chunks of 128; weights stream global->LDS via async
// copies pipelined one chunk ahead; FFN2 accumulates fp32 in registers.
// ---------------------------------------------------------------------------
__global__ __launch_bounds__(512) void mhff_kernel(
    const float* __restrict__ x,
    const __bf16* __restrict__ W1t, const float* __restrict__ b1,
    const __bf16* __restrict__ W2t, const float* __restrict__ b2,
    float* __restrict__ out) {
  extern __shared__ char smem[];
  __bf16* hA   = (__bf16*)smem;                 // [128][128] tokens x f-chunk
  __bf16* w2c  = (__bf16*)(smem + 32 * 1024);   // [128 d][128 f] chunk of W2t
  __bf16* w1c0 = (__bf16*)(smem + 64 * 1024);   // [128 f][128 d] chunk of W1t
  __bf16* w1c1 = (__bf16*)(smem + 96 * 1024);

  const int head = blockIdx.y;
  const int t0 = blockIdx.x * M_BLK;
  const int tid = threadIdx.x;
  const int wave = tid >> 5, lane = tid & 31;
  const int lmod = lane & 15, lg = lane >> 4;
  const int m = wave & 7;    // wave's 16-token row tile (0..7)
  const int ng = wave >> 3;  // wave's n-tile parity (0..1)

  const __bf16* W1h = W1t + head * HEAD_DIM * FF_DIM;
  const __bf16* W2h = W2t + head * HEAD_DIM * FF_DIM;
  const float* b1h = b1 + head * FF_DIM;

  // ---- A-fragments for FFN1 straight from global: per lane 8+8 (at k 8*lg and 16+8*lg)
  // floats of token row (t0 + m*16 + lane%16), packed to bf16. Built once. ----
  Frag a1[4];
#pragma unroll
  for (int k = 0; k < 4; ++k) {
    const float* p = x + (size_t)(t0 + m * 16 + lmod) * D_MODEL +
                     head * HEAD_DIM + k * 32 + lg * 8;
    float4 f0 = *(const float4*)p;
    float4 f1 = *(const float4*)(p + 4);
    float4 f2 = *(const float4*)(p + 16);
    float4 f3 = *(const float4*)(p + 20);
    v16bf a;
    a[0] = (__bf16)f0.x;  a[1] = (__bf16)f0.y;  a[2] = (__bf16)f0.z;  a[3] = (__bf16)f0.w;
    a[4] = (__bf16)f1.x;  a[5] = (__bf16)f1.y;  a[6] = (__bf16)f1.z;  a[7] = (__bf16)f1.w;
    a[8] = (__bf16)f2.x;  a[9] = (__bf16)f2.y;  a[10] = (__bf16)f2.z; a[11] = (__bf16)f2.w;
    a[12] = (__bf16)f3.x; a[13] = (__bf16)f3.y; a[14] = (__bf16)f3.z; a[15] = (__bf16)f3.w;
    a1[k].v = a;
  }

  // ---- prologue: stage W1 chunk 0 (contiguous 32KB), 4 x 16B per thread ----
#pragma unroll
  for (int i = 0; i < 4; ++i) {
    int p = tid + i * 512;  // 2048 pieces of 16B
    stage16(W1h + p * 8, w1c0 + p * 8);
  }
  wait_stage();
  __syncthreads();

  const v8f zero = {};
  v8f c2[4];
#pragma unroll
  for (int j = 0; j < 4; ++j) c2[j] = zero;

  for (int fc = 0; fc < FF_DIM / F_CHUNK; ++fc) {
    __bf16* w1cur = (fc & 1) ? w1c1 : w1c0;
    __bf16* w1nxt = (fc & 1) ? w1c0 : w1c1;

    // ---- issue async stages: W1 chunk fc+1, W2 chunk fc (hidden by FFN1) ----
    if (fc < 3) {
#pragma unroll
      for (int i = 0; i < 4; ++i) {
        int p = tid + i * 512;
        stage16(W1h + (fc + 1) * (F_CHUNK * HEAD_DIM) + p * 8, w1nxt + p * 8);
      }
    }
#pragma unroll
    for (int i = 0; i < 4; ++i) {
      int p = tid + i * 512;       // row = p>>4 (d), seg = p&15 (16B of f)
      int row = p >> 4, seg = p & 15;
      stage16(W2h + row * FF_DIM + fc * F_CHUNK + seg * 8,
              w2c + row * F_CHUNK + seg * 8);
    }

    // ---- FFN1: hA = relu(x @ W1[:, chunk] + b1[chunk]), B-frags from LDS ----
#pragma unroll
    for (int j = 0; j < 4; ++j) {
      const int nn = ng + 2 * j;  // n-tile within chunk, 0..7
      v8f c = zero;
#pragma unroll
      for (int k = 0; k < 4; ++k) {
        const __bf16* bp = w1cur + (nn * 16 + lmod) * HEAD_DIM + k * 32 + lg * 8;
        Frag b;
        b.h[0] = *(const v8bf*)bp;
        b.h[1] = *(const v8bf*)(bp + 16);
        c = wmma_bf16(a1[k].v, b.v, c);
      }
      const float bias = b1h[fc * F_CHUNK + nn * 16 + lmod];
      // C layout: VGPR r, lanes 0-15: M=r, N=lane; lanes 16-31: M=8+r
#pragma unroll
      for (int r = 0; r < 8; ++r) {
        float v = c[r] + bias;
        v = v > 0.0f ? v : 0.0f;
        hA[(m * 16 + r + lg * 8) * F_CHUNK + nn * 16 + lmod] = (__bf16)v;
      }
    }
    wait_stage();     // W2(fc) [and W1(fc+1)] resident
    __syncthreads();  // hA visible to all waves

    // ---- FFN2 partial: c2 += relu_h_chunk @ W2[chunk, :], all from LDS ----
    Frag a2[4];
#pragma unroll
    for (int k = 0; k < 4; ++k) {
      const __bf16* p = hA + (m * 16 + lmod) * F_CHUNK + k * 32 + lg * 8;
      a2[k].h[0] = *(const v8bf*)p;
      a2[k].h[1] = *(const v8bf*)(p + 16);
    }
#pragma unroll
    for (int j = 0; j < 4; ++j) {
      const int n2 = ng + 2 * j;  // out-dim n-tile, 0..7
#pragma unroll
      for (int k = 0; k < 4; ++k) {
        const __bf16* bp = w2c + (n2 * 16 + lmod) * F_CHUNK + k * 32 + lg * 8;
        Frag b;
        b.h[0] = *(const v8bf*)bp;
        b.h[1] = *(const v8bf*)(bp + 16);
        c2[j] = wmma_bf16(a2[k].v, b.v, c2[j]);
      }
    }
    __syncthreads();  // protect hA / w2c before next chunk's writes
  }

  // ---- epilogue: + b2, fp32 store ----
  const float* b2h = b2 + head * HEAD_DIM;
#pragma unroll
  for (int j = 0; j < 4; ++j) {
    const int n2 = ng + 2 * j;
    const float bias = b2h[n2 * 16 + lmod];
    float* op = out + (size_t)(t0 + m * 16 + lg * 8) * D_MODEL +
                head * HEAD_DIM + n2 * 16 + lmod;
#pragma unroll
    for (int r = 0; r < 8; ++r) op[(size_t)r * D_MODEL] = c2[j][r] + bias;
  }
}

extern "C" void kernel_launch(void* const* d_in, const int* in_sizes, int n_in,
                              void* d_out, int out_size, void* d_ws,
                              size_t ws_size, hipStream_t stream) {
  (void)in_sizes; (void)n_in; (void)out_size; (void)ws_size;
  const float* x  = (const float*)d_in[0];
  const float* W1 = (const float*)d_in[1];
  const float* b1 = (const float*)d_in[2];
  const float* W2 = (const float*)d_in[3];
  const float* b2 = (const float*)d_in[4];
  float* out = (float*)d_out;

  // workspace: bf16 transposed weights (2 x 2MB = 4MB)
  __bf16* W1t = (__bf16*)d_ws;
  __bf16* W2t = W1t + N_HEADS * HEAD_DIM * FF_DIM;

  cvt_w_kernel<<<1024, 256, 0, stream>>>(W1, W2, W1t, W2t);

  dim3 grid(TOKENS / M_BLK, N_HEADS);
  mhff_kernel<<<grid, 512, 128 * 1024, stream>>>(x, W1t, b1, W2t, b2, out);
}
